// MultiHeadAttention_32341103739229
// MI455X (gfx1250) — matrix operand order repaired and checked by reading, not compiled, not run on hardware
//
#include <hip/hip_runtime.h>
#include <hip/hip_bf16.h>

typedef __bf16 bf16_t;
typedef __attribute__((ext_vector_type(16))) __bf16 v16bf;
typedef __attribute__((ext_vector_type(8)))  __bf16 bf16x8;
typedef __attribute__((ext_vector_type(8)))  float  v8f;

#define H_HEADS 8
#define D_HID   512
#define N_BATCH 4
#define SEQ     1024
#define DH      64
#define ROWS    (N_BATCH * SEQ)   // 4096
#define NEGBIG  (-1.0e9f)

#define WMMA_BF16(A, B, C) \
  __builtin_amdgcn_wmma_f32_16x16x32_bf16(false, (A), false, (B), (short)0, (C), false, false)

// concat two 8-wide bf16 vectors into a 16-wide WMMA operand
static __device__ inline v16bf cat8(bf16x8 a, bf16x8 b) {
  return __builtin_shufflevector(a, b, 0,1,2,3,4,5,6,7,8,9,10,11,12,13,14,15);
}

// convert 8 consecutive f32 (16B aligned) to bf16x8
static __device__ inline bf16x8 cvt8(const float* p) {
  const float4* q = (const float4*)p;
  float4 a = q[0], b = q[1];
  bf16x8 r;
  r[0] = (__bf16)a.x; r[1] = (__bf16)a.y; r[2] = (__bf16)a.z; r[3] = (__bf16)a.w;
  r[4] = (__bf16)b.x; r[5] = (__bf16)b.y; r[6] = (__bf16)b.z; r[7] = (__bf16)b.w;
  return r;
}

// load 16 contiguous bf16 (32B, 16B aligned) as an operand chunk
static __device__ inline v16bf load16(const bf16_t* p) {
  const bf16x8* q = (const bf16x8*)p;
  return cat8(q[0], q[1]);
}

// reductions across the 16-lane half-wave (row of a C-layout tile)
static __device__ inline float hmax16(float x) {
  x = fmaxf(x, __shfl_xor(x, 1, 32));
  x = fmaxf(x, __shfl_xor(x, 2, 32));
  x = fmaxf(x, __shfl_xor(x, 4, 32));
  x = fmaxf(x, __shfl_xor(x, 8, 32));
  return x;
}
static __device__ inline float hsum16(float x) {
  x += __shfl_xor(x, 1, 32);
  x += __shfl_xor(x, 2, 32);
  x += __shfl_xor(x, 4, 32);
  x += __shfl_xor(x, 8, 32);
  return x;
}

// one per-lane async 16B global->LDS transfer (ASYNCcnt-tracked)
static __device__ inline void async_b128(unsigned lds_off, unsigned long long gaddr) {
  asm volatile("global_load_async_to_lds_b128 %0, %1, off"
               :: "v"(lds_off), "v"(gaddr) : "memory");
}

// ---------------------------------------------------------------------------
// Stage 1: Y[4096,512] = X @ W^T + b  (f32 in, bf16 out)
// mode 0: head-split [m, s, dh]   (for Q, K)
// mode 1: per-head transposed [m, dh, s]  (for V, so PV B-operand is contiguous)
// One wave computes a 16x64 tile (4 WMMA accumulators, A reused 4x).
// ---------------------------------------------------------------------------
__global__ void proj_bf16_kernel(const float* __restrict__ X,
                                 const float* __restrict__ W,
                                 const float* __restrict__ bias,
                                 bf16_t* __restrict__ out, int mode) {
  const int lane = threadIdx.x;
  const int lo = lane & 15, hi = lane >> 4;
  const int r0 = blockIdx.x * 16;
  const int c0 = blockIdx.y * 64;

  v8f acc[4] = {};
  const float* xr = X + (size_t)(r0 + lo) * D_HID;

  for (int k0 = 0; k0 < D_HID; k0 += 32) {
    if (k0 + 32 < D_HID) __builtin_prefetch(xr + k0 + 32, 0, 1);
    // A operand 16x32: lane=row, K = k0 + 16*hi + 0..15 (contiguous)
    v16bf a = cat8(cvt8(xr + k0 + 16 * hi), cvt8(xr + k0 + 16 * hi + 8));
#pragma unroll
    for (int t = 0; t < 4; ++t) {
      // B[k][n] = W[n][k]; lane=col n, K = k0 + 16*hi + h (16 contiguous f32)
      const float* wr = W + (size_t)(c0 + 16 * t + lo) * D_HID + k0 + 16 * hi;
      v16bf b = cat8(cvt8(wr), cvt8(wr + 8));
      acc[t] = WMMA_BF16(a, b, acc[t]);
    }
  }

#pragma unroll
  for (int t = 0; t < 4; ++t) {
    const int col = c0 + 16 * t + lo;
    const float bb = bias[col];
    const int h = col >> 6, c = col & (DH - 1);
#pragma unroll
    for (int v = 0; v < 8; ++v) {
      const int row = r0 + v + 8 * hi;
      const int n = row >> 10, q = row & (SEQ - 1);
      const int m = n * H_HEADS + h;
      const size_t idx = (mode == 0)
          ? ((size_t)(m * SEQ + q)) * DH + c
          : ((size_t)(m * DH + c)) * SEQ + q;
      out[idx] = (bf16_t)(acc[t][v] + bb);
    }
  }
}

// ---------------------------------------------------------------------------
// Stage 2: flash attention. One wave per (head m, 16-query tile).
// K/V blocks staged to LDS with double-buffered async copies; online softmax;
// P converted C-layout -> A-layout via an LDS bounce.
// ---------------------------------------------------------------------------

// stage one 32-key block: K tile (contiguous 4KB) + V tile (64 rows x 64B)
static __device__ inline void stage_kv(const bf16_t* __restrict__ Kb,
                                       const bf16_t* __restrict__ Vt,
                                       int m, int s0, int lane,
                                       bf16_t* kdst_p, bf16_t* vdst_p) {
  const unsigned kdst = (unsigned)(unsigned long long)(const void*)kdst_p;
  const unsigned vdst = (unsigned)(unsigned long long)(const void*)vdst_p;
  const unsigned long long kg =
      (unsigned long long)(Kb + ((size_t)(m * SEQ + s0)) * DH);
#pragma unroll
  for (int j = 0; j < 8; ++j) {           // 8 x (32 lanes x 16B) = 4KB
    const unsigned off = (unsigned)(lane + 32 * j) * 16u;
    async_b128(kdst + off, kg + off);
  }
#pragma unroll
  for (int j = 0; j < 8; ++j) {           // 64 rows x 64B, 16B pieces
    const int t = lane + 32 * j;
    const int c = t >> 2, part = t & 3;
    const unsigned long long src =
        (unsigned long long)(Vt + ((size_t)(m * DH + c)) * SEQ + s0 + part * 8);
    async_b128(vdst + (unsigned)(c * 64 + part * 16), src);
  }
}

__global__ void attn_kernel(const bf16_t* __restrict__ Qb,
                            const bf16_t* __restrict__ Kb,
                            const bf16_t* __restrict__ Vt,
                            const int* __restrict__ prefix,
                            bf16_t* __restrict__ attn) {
  const int lane = threadIdx.x;
  const int lo = lane & 15, hi = lane >> 4;
  const int m = blockIdx.y;
  const int q0 = blockIdx.x * 16;
  const int pfx = prefix[m >> 3];   // prefix repeated per head, n-major

  __shared__ __attribute__((aligned(16))) bf16_t kbuf[2][32 * DH];  // [key][dh]
  __shared__ __attribute__((aligned(16))) bf16_t vbuf[2][DH * 32];  // [dh][key]
  __shared__ __attribute__((aligned(16))) bf16_t lds_p[16 * 32];

  // Q A-operands for K-dim = dh = 64 (two 32-wide chunks), reused all blocks
  const bf16_t* qrow = Qb + ((size_t)(m * SEQ + q0 + lo)) * DH;
  const v16bf aq0 = cat8(*(const bf16x8*)(qrow + 16 * hi),
                         *(const bf16x8*)(qrow + 16 * hi + 8));
  const v16bf aq1 = cat8(*(const bf16x8*)(qrow + 32 + 16 * hi),
                         *(const bf16x8*)(qrow + 32 + 16 * hi + 8));

  float mrun[8], lrun[8];
  v8f acc[4] = {};
#pragma unroll
  for (int v = 0; v < 8; ++v) { mrun[v] = -3.0e38f; lrun[v] = 0.0f; }

  // prime the pipeline: block 0 in flight
  stage_kv(Kb, Vt, m, 0, lane, kbuf[0], vbuf[0]);

  for (int blk = 0; blk < SEQ / 32; ++blk) {
    const int s0 = blk * 32;
    const int buf = blk & 1;

    if (blk + 1 < SEQ / 32) {
      // buffer buf^1 was last read in the previous iteration; drain LDS reads first
      asm volatile("s_wait_dscnt 0x0" ::: "memory");
      stage_kv(Kb, Vt, m, s0 + 32, lane, kbuf[buf ^ 1], vbuf[buf ^ 1]);
      // wait for current block (16 older transfers) while next 16 stay in flight
      asm volatile("s_wait_asynccnt 0x10" ::: "memory");
    } else {
      asm volatile("s_wait_asynccnt 0x0" ::: "memory");
    }
    __syncthreads();

    const bf16_t* kb0 = kbuf[buf];
    const bf16_t* vb0 = vbuf[buf];

    // --- scores for 32 keys: two 16x16 tiles, K=64 via 2 WMMAs each ---
    v8f sA = {}, sB = {};
    {
      const bf16_t* kr = kb0 + lo * DH + 16 * hi;         // keys 0..15
      sA = WMMA_BF16(aq0, load16(kr), sA);
      sA = WMMA_BF16(aq1, load16(kr + 32), sA);
      const bf16_t* kr2 = kb0 + (16 + lo) * DH + 16 * hi; // keys 16..31
      sB = WMMA_BF16(aq0, load16(kr2), sB);
      sB = WMMA_BF16(aq1, load16(kr2 + 32), sB);
    }

    // --- mask + online softmax (rows live in 16-lane halves) ---
    const int sidxA = s0 + lo, sidxB = sidxA + 16;
#pragma unroll
    for (int v = 0; v < 8; ++v) {
      const int qi = q0 + v + 8 * hi;
      const float xa = ((sidxA < pfx) || (sidxA == qi)) ? sA[v] * 0.125f : NEGBIG;
      const float xb = ((sidxB < pfx) || (sidxB == qi)) ? sB[v] * 0.125f : NEGBIG;
      const float bm   = hmax16(fmaxf(xa, xb));
      const float mnew = fmaxf(mrun[v], bm);
      const float alpha = __expf(mrun[v] - mnew);
      const float pa = __expf(xa - mnew);
      const float pb = __expf(xb - mnew);
      const float rs = hsum16(pa + pb);
      lrun[v] = lrun[v] * alpha + rs;
      mrun[v] = mnew;
      acc[0][v] *= alpha; acc[1][v] *= alpha;
      acc[2][v] *= alpha; acc[3][v] *= alpha;
      const int r = v + 8 * hi;            // row of the P tile this lane holds
      lds_p[r * 32 + lo]      = (bf16_t)pa;
      lds_p[r * 32 + 16 + lo] = (bf16_t)pb;
    }
    __syncthreads();

    // --- re-read P as a 16x32 A operand ---
    const v16bf ap = cat8(*(const bf16x8*)(lds_p + lo * 32 + 16 * hi),
                          *(const bf16x8*)(lds_p + lo * 32 + 16 * hi + 8));

    // --- acc += P @ V ; vbuf is [dh][key] so B operand is contiguous ---
#pragma unroll
    for (int t = 0; t < 4; ++t) {
      const bf16_t* vr = vb0 + (16 * t + lo) * 32 + 16 * hi;
      acc[t] = WMMA_BF16(ap, load16(vr), acc[t]);
    }
    __syncthreads();
  }

  // normalize and store merged-head bf16 [row=n*S+q, col=h*dh+c]
  const int n = m >> 3, h = m & (H_HEADS - 1);
#pragma unroll
  for (int t = 0; t < 4; ++t) {
    const int c = 16 * t + lo;
#pragma unroll
    for (int v = 0; v < 8; ++v) {
      const int q = q0 + v + 8 * hi;
      const float o = acc[t][v] / lrun[v];
      attn[((size_t)(n * SEQ + q)) * D_HID + h * DH + c] = (bf16_t)o;
    }
  }
}

// ---------------------------------------------------------------------------
// Stage 3: out[4096,512] f32 = attn_bf16 @ Wo^T + bo
// ---------------------------------------------------------------------------
__global__ void out_proj_kernel(const bf16_t* __restrict__ Xb,
                                const float* __restrict__ W,
                                const float* __restrict__ bias,
                                float* __restrict__ out) {
  const int lane = threadIdx.x;
  const int lo = lane & 15, hi = lane >> 4;
  const int r0 = blockIdx.x * 16;
  const int c0 = blockIdx.y * 64;

  v8f acc[4] = {};
  const bf16_t* xr = Xb + (size_t)(r0 + lo) * D_HID;

  for (int k0 = 0; k0 < D_HID; k0 += 32) {
    v16bf a = cat8(*(const bf16x8*)(xr + k0 + 16 * hi),
                   *(const bf16x8*)(xr + k0 + 16 * hi + 8));
#pragma unroll
    for (int t = 0; t < 4; ++t) {
      const float* wr = W + (size_t)(c0 + 16 * t + lo) * D_HID + k0 + 16 * hi;
      v16bf b = cat8(cvt8(wr), cvt8(wr + 8));
      acc[t] = WMMA_BF16(a, b, acc[t]);
    }
  }

#pragma unroll
  for (int t = 0; t < 4; ++t) {
    const int col = c0 + 16 * t + lo;
    const float bb = bias[col];
#pragma unroll
    for (int v = 0; v < 8; ++v) {
      const int row = r0 + v + 8 * hi;
      out[(size_t)row * D_HID + col] = acc[t][v] + bb;
    }
  }
}

// ---------------------------------------------------------------------------
extern "C" void kernel_launch(void* const* d_in, const int* in_sizes, int n_in,
                              void* d_out, int out_size, void* d_ws, size_t ws_size,
                              hipStream_t stream) {
  const float* Xq = (const float*)d_in[0];
  const float* Xk = (const float*)d_in[1];
  const float* Xv = (const float*)d_in[2];
  const int*  pfx = (const int*)d_in[3];
  const float* Wq = (const float*)d_in[4];
  const float* bq = (const float*)d_in[5];
  const float* Wk = (const float*)d_in[6];
  const float* bk = (const float*)d_in[7];
  const float* Wv = (const float*)d_in[8];
  const float* bv = (const float*)d_in[9];
  const float* Wo = (const float*)d_in[10];
  const float* bo = (const float*)d_in[11];
  float* out = (float*)d_out;

  // scratch carve-up: 4 bf16 arrays of 32*1024*64 = 2M elements (4 MB each)
  const size_t E = (size_t)32 * SEQ * DH;
  bf16_t* qb = (bf16_t*)d_ws;
  bf16_t* kb = qb + E;
  bf16_t* vt = kb + E;
  bf16_t* at = vt + E;

  dim3 gP(ROWS / 16, D_HID / 64);   // 256 x 8 waves
  proj_bf16_kernel<<<gP, 32, 0, stream>>>(Xq, Wq, bq, qb, 0);
  proj_bf16_kernel<<<gP, 32, 0, stream>>>(Xk, Wk, bk, kb, 0);
  proj_bf16_kernel<<<gP, 32, 0, stream>>>(Xv, Wv, bv, vt, 1);
  attn_kernel<<<dim3(SEQ / 16, 32), 32, 0, stream>>>(qb, kb, vt, pfx, at);
  out_proj_kernel<<<gP, 32, 0, stream>>>(at, Wo, bo, out);
}
